// TrIPModel_44787918963222
// MI455X (gfx1250) — matrix operand order repaired and checked by reading, not compiled, not run on hardware
//
#include <hip/hip_runtime.h>
#include <hip/hip_bf16.h>
#include <math.h>

typedef __attribute__((ext_vector_type(16))) _Float16 v16h_t;
typedef __attribute__((ext_vector_type(8)))  float    v8f_t;

#define NN 32768
#define EE 524288
#define CCH 16
#define LLAY 4
#define KCH 8
#define CUTOFF_F 4.6f
#define Y0_F 0.28209479177387814f
#define NWAVE 8

// ---------- node init: x = emb[species-1]; zero feats/coul/out ----------
__global__ void k_node_init(const int* __restrict__ species, const float* __restrict__ emb,
                            float* x, float* feats, float* coul, float* out) {
  int n = blockIdx.x * blockDim.x + threadIdx.x;
  if (n == 0) out[0] = 0.f;
  if (n < NN) {
    int sp = species[n] - 1;
#pragma unroll
    for (int i = 0; i < CCH; i++) { x[n * CCH + i] = emb[sp * CCH + i]; feats[n * CCH + i] = 0.f; }
    coul[n] = 0.f;
  }
}

// ---------- edge setup: dist, cutoff envelope, ZBL coulomb ----------
__global__ void k_edge_setup(const int* __restrict__ species, const int* __restrict__ src,
                             const int* __restrict__ dst, const float* __restrict__ rel,
                             float* distv, float* scalev, float* bscalev, float* coul) {
  int e = blockIdx.x * blockDim.x + threadIdx.x;
  if (e >= EE) return;
  float dx = rel[e * 3], dy = rel[e * 3 + 1], dz = rel[e * 3 + 2];
  float d = sqrtf(dx * dx + dy * dy + dz * dz);
  distv[e] = d;
  float sc = 0.f;
  if (d < CUTOFF_F) {
    float xc = d / CUTOFF_F;
    float x2 = fminf(xc * xc, 1.f - 1e-6f);
    sc = __expf(3.f - 3.f / (1.f - x2));
  }
  scalev[e] = sc;
  bscalev[e] = Y0_F * sc;
  int s = src[e], t = dst[e];
  float Zu = (float)species[s], Zv = (float)species[t];
  float raw = 0.529f * Zu * Zv / (2.f * d);
  float au = 0.8854f * 0.529f / (powf(Zu, 0.23f) + powf(Zv, 0.23f));
  float xx = d / au;
  float screen = 0.1818f * __expf(-3.2f * xx) + 0.5099f * __expf(-0.9423f * xx)
               + 0.2802f * __expf(-0.4028f * xx) + 0.02817f * __expf(-0.2016f * xx);
  atomicAdd(&coul[t], raw * screen * sc);
}

// ---------- per-layer: q = x @ Wq^T ; zero numer/denom ----------
__global__ void k_q(const float* __restrict__ x, const float* __restrict__ Wq,
                    float* q, float* numer, float* denom) {
  int n = blockIdx.x * blockDim.x + threadIdx.x;
  if (n >= NN) return;
  float xr[CCH];
#pragma unroll
  for (int i = 0; i < CCH; i++) xr[i] = x[n * CCH + i];
#pragma unroll
  for (int h = 0; h < KCH; h++) {
    float acc = 0.f;
#pragma unroll
    for (int i = 0; i < CCH; i++) acc += xr[i] * Wq[h * CCH + i];
    q[n * KCH + h] = acc;
    numer[n * KCH + h] = 0.f;
    denom[n * KCH + h] = 0.f;
  }
}

// ---------- WMMA edge kernel: radial MLP + einsum (all on matrix pipe) ----------
// mode 0: layer 0 (ef from dist, writeback). mode 1: layers 1..3 (ef += x[src], writeback).
// mode 2: final conv (ef += x[src], feats atomics instead of attention).
__global__ __launch_bounds__(256, 1)
void k_edge(int mode,
            const int* __restrict__ src, const int* __restrict__ dst,
            const float* __restrict__ distv, const float* __restrict__ scalev,
            const float* __restrict__ bscalev,
            const float* __restrict__ x, const float* __restrict__ q, float* efws,
            const float* __restrict__ W1, const float* __restrict__ B1,
            const float* __restrict__ W2, const float* __restrict__ B2,
            const float* __restrict__ W3, const float* __restrict__ B3,
            float* numer, float* denom, float* feats) {
  // weights pre-swizzled into B-fragment order: sW[t*32 + lane] is the v16h for that lane
  __shared__ v16h_t sW1s[2 * 32];
  __shared__ v16h_t sW2s[2 * 32];
  __shared__ v16h_t sW3s[16 * 32];
  __shared__ float sB1[32], sB2[32], sB3[256];
  // per-wave staging: A-fragment-order f16 buffers + f32 helpers
  __shared__ v16h_t sEFa[NWAVE][32];
  __shared__ v16h_t sHa [NWAVE][32];
  __shared__ float sKV[NWAVE][16 * 16];
  __shared__ float sQD[NWAVE][16 * 8];
  __shared__ float sSc[NWAVE][16];
  __shared__ float sBs[NWAVE][16];
  __shared__ int   sDs[NWAVE][16];

  const int tid = threadIdx.x;
  // ---- stage weights, swizzled to B-fragment layout: idx = (t*32 + lanec)*16 + j,
  //      element = W[K][t*16+n] with K = j + 16*(lanec>>4), n = lanec&15 ----
  {
    _Float16* wh = (_Float16*)sW1s;
    for (int idx = tid; idx < 1024; idx += blockDim.x) {
      int t = idx >> 9, rem = idx & 511, lanec = rem >> 4, j = rem & 15;
      int K = j + 16 * (lanec >> 4), col = t * 16 + (lanec & 15);
      wh[idx] = (K < 16) ? (_Float16)W1[K * 32 + col] : (_Float16)0.f;  // K=16..31 padded
    }
    wh = (_Float16*)sW2s;
    for (int idx = tid; idx < 1024; idx += blockDim.x) {
      int t = idx >> 9, rem = idx & 511, lanec = rem >> 4, j = rem & 15;
      int K = j + 16 * (lanec >> 4), col = t * 16 + (lanec & 15);
      wh[idx] = (_Float16)W2[K * 32 + col];
    }
    wh = (_Float16*)sW3s;
    for (int idx = tid; idx < 8192; idx += blockDim.x) {
      int t = idx >> 9, rem = idx & 511, lanec = rem >> 4, j = rem & 15;
      int K = j + 16 * (lanec >> 4), col = t * 16 + (lanec & 15);
      wh[idx] = (_Float16)W3[K * 256 + col];
    }
    for (int i = tid; i < 32; i += blockDim.x) { sB1[i] = B1[i]; sB2[i] = B2[i]; }
    for (int i = tid; i < 256; i += blockDim.x) sB3[i] = B3[i];
  }

  const int w = tid >> 5, lane = tid & 31;
  const int hi = lane >> 4, lo = lane & 15;
  const long ebase = ((long)blockIdx.x * NWAVE + w) * 16;

  // zero A-frag buffer (covers the K>=16 padding halves of GEMM1's A)
  {
    v16h_t z = {};
    sEFa[w][lane] = z;
  }
  __syncthreads();

  // ---- per-edge loads: lanes 0..15 own one edge each (vectorized float4) ----
  // bx = register-resident B fragment of xs^T: B2[i][n] = xs[n][i] (lane n holds column n)
  v16h_t bx = {};
  if (lane < 16) {
    long e = ebase + lane;
    int s = src[e], d = dst[e];
    sDs[w][lane] = d; sSc[w][lane] = scalev[e]; sBs[w][lane] = bscalev[e];
    float de = distv[e];
    const float4* x4 = (const float4*)x;
    float4* ef4 = (float4*)efws;
    _Float16* efh = (_Float16*)&sEFa[w][0];
#pragma unroll
    for (int k = 0; k < 4; k++) {
      float4 xs = x4[(long)s * 4 + k];
      float xv[4] = {xs.x, xs.y, xs.z, xs.w};
#pragma unroll
      for (int kk = 0; kk < 4; kk++) bx[k * 4 + kk] = (_Float16)xv[kk];
      float4 ev;
      if (mode == 0) {
        ev.x = 0.f; ev.y = 0.f; ev.z = 0.f;
        ev.w = (k == 3) ? de : 0.f;
      } else {
        float4 eo = ef4[e * 4 + k];
        ev.x = eo.x + xs.x; ev.y = eo.y + xs.y; ev.z = eo.z + xs.z; ev.w = eo.w + xs.w;
      }
      if (mode <= 1) ef4[e * 4 + k] = ev;
      // scatter into A-fragment layout: feature i -> lane' = lane, half j = i
      float evs[4] = {ev.x, ev.y, ev.z, ev.w};
#pragma unroll
      for (int kk = 0; kk < 4; kk++) {
        int i = k * 4 + kk;
        efh[lane * 16 + i] = (_Float16)evs[kk];
      }
    }
    if (mode < 2) {
      const float4* q4 = (const float4*)q;
#pragma unroll
      for (int k = 0; k < 2; k++)
        *(float4*)&sQD[w][lane * 8 + k * 4] = q4[(long)d * 2 + k];
    }
  }
  __syncthreads();

  _Float16* hh = (_Float16*)&sHa[w][0];

  // ---- GEMM1: h1 = relu(ef @ W1 + b1)  (K=16 padded to 32, N=32) ----
  {
    v16h_t a = sEFa[w][lane];
#pragma unroll
    for (int nt = 0; nt < 2; nt++) {
      v16h_t b = sW1s[nt * 32 + lane];
      v8f_t acc; float bv = sB1[nt * 16 + lo];
#pragma unroll
      for (int r = 0; r < 8; r++) acc[r] = bv;
      acc = __builtin_amdgcn_wmma_f32_16x16x32_f16(false, a, false, b, (short)0, acc, false, false);
      // D element (M=r+8*hi, c=nt*16+lo) -> next-A slot lane' = nt*16 + M, half lo
#pragma unroll
      for (int r = 0; r < 8; r++)
        hh[(nt * 16 + r + 8 * hi) * 16 + lo] =
            (_Float16)fmaxf(acc[r], 0.f);
    }
  }
  __syncthreads();

  // ---- GEMM2: h2 = relu(h1 @ W2 + b2)  (K=32, N=32), in place in sHa ----
  {
    v16h_t a = sHa[w][lane];
    __syncthreads();  // everyone captured h1 before overwrite
#pragma unroll
    for (int nt = 0; nt < 2; nt++) {
      v16h_t b = sW2s[nt * 32 + lane];
      v8f_t acc; float bv = sB2[nt * 16 + lo];
#pragma unroll
      for (int r = 0; r < 8; r++) acc[r] = bv;
      acc = __builtin_amdgcn_wmma_f32_16x16x32_f16(false, a, false, b, (short)0, acc, false, false);
#pragma unroll
      for (int r = 0; r < 8; r++)
        hh[(nt * 16 + r + 8 * hi) * 16 + lo] =
            (_Float16)fmaxf(acc[r], 0.f);
    }
  }
  __syncthreads();

  // ---- GEMM3: R = h2 @ W3 + b3 (16 N-tiles); einsum kv[e][t] = diag(R_t @ xs^T) via WMMA ----
  {
    v16h_t a = sHa[w][lane];       // h2 fragment, captured
    __syncthreads();
    { v16h_t z = {}; sHa[w][lane] = z; }  // reuse sHa as R-frag staging; zero K>=16 padding
#pragma unroll 4
    for (int t = 0; t < 16; t++) {
      v16h_t b = sW3s[t * 32 + lane];
      v8f_t acc; float bv = sB3[t * 16 + lo];
#pragma unroll
      for (int r = 0; r < 8; r++) acc[r] = bv;
      acc = __builtin_amdgcn_wmma_f32_16x16x32_f16(false, a, false, b, (short)0, acc, false, false);
      // acc[r] = R[e = r+8*hi][i = lo]; scatter to A-frag slot (lane'=M, j=lo)
#pragma unroll
      for (int r = 0; r < 8; r++)
        hh[(r + 8 * hi) * 16 + lo] = (_Float16)acc[r];
      v16h_t a2 = sHa[w][lane];    // same-wave in-order LDS: stores above are visible
      v8f_t acc2;
#pragma unroll
      for (int r = 0; r < 8; r++) acc2[r] = 0.f;
      acc2 = __builtin_amdgcn_wmma_f32_16x16x32_f16(false, a2, false, bx, (short)0, acc2, false, false);
      // result2[M][N] = sum_i R[M][i]*xs[N][i]; diag M==N==e -> kv[e][t]
      if (hi == (lo >> 3)) {
        int rr = lo & 7;
        float dv = acc2[0];
#pragma unroll
        for (int r = 1; r < 8; r++) dv = (rr == r) ? acc2[r] : dv;
        sKV[w][lo * 16 + t] = dv;
      }
    }
  }
  __syncthreads();

  // ---- epilogue: attention numer/denom (mode<2) or feats segment-sum (mode 2) ----
  if (lane < 16) {
    int e = lane;
    int d = sDs[w][e];
    float sc = sSc[w][e], bs = sBs[w][e];
    if (mode < 2) {
#pragma unroll
      for (int h = 0; h < 8; h++) {
        float valv = sKV[w][e * 16 + h] * bs;
        float keyv = sKV[w][e * 16 + 8 + h] * bs;
        float logit = keyv * sQD[w][e * 8 + h];       // dh == 1, /sqrt(1)
        float wt = sc * __expf(fminf(logit, 80.f));   // softmax max-shift cancels in ratio
        atomicAdd(&numer[(long)d * 8 + h], wt * valv);
        atomicAdd(&denom[(long)d * 8 + h], wt);
      }
    } else {
#pragma unroll
      for (int o = 0; o < 16; o++)
        atomicAdd(&feats[(long)d * 16 + o], sKV[w][e * 16 + o] * bs);
    }
  }
}

// ---------- per-layer node update: x = [numer/denom, x] @ Wproj^T ----------
__global__ void k_node_update(const float* __restrict__ numer, const float* __restrict__ denom,
                              const float* __restrict__ Wproj, float* x) {
  __shared__ float sWp[16 * 24];
  for (int i = threadIdx.x; i < 16 * 24; i += blockDim.x) sWp[i] = Wproj[i];
  __syncthreads();
  int n = blockIdx.x * blockDim.x + threadIdx.x;
  if (n >= NN) return;
  float cat[24];
#pragma unroll
  for (int h = 0; h < 8; h++) cat[h] = numer[n * 8 + h] / fmaxf(denom[n * 8 + h], 1e-20f);
#pragma unroll
  for (int i = 0; i < 16; i++) cat[8 + i] = x[n * 16 + i];
  float xn[16];
#pragma unroll
  for (int c = 0; c < 16; c++) {
    float acc = 0.f;
#pragma unroll
    for (int k = 0; k < 24; k++) acc += cat[k] * sWp[c * 24 + k];
    xn[c] = acc;
  }
#pragma unroll
  for (int c = 0; c < 16; c++) x[n * 16 + c] = xn[c];
}

// ---------- final node MLP + coulomb + global reduction ----------
__global__ void k_node_final(const int* __restrict__ species, const float* __restrict__ emb,
                             const float* __restrict__ x, const float* __restrict__ feats,
                             const float* __restrict__ coul,
                             const float* __restrict__ Wself,
                             const float* __restrict__ mW1, const float* __restrict__ mb1,
                             const float* __restrict__ mW2, const float* __restrict__ mb2,
                             const float* __restrict__ mW3, const float* __restrict__ mb3,
                             float* out) {
  __shared__ float sWs[256], sM1[512], sM2[256], sM3[16];
  __shared__ float red[256];
  for (int i = threadIdx.x; i < 256; i += blockDim.x) { sWs[i] = Wself[i]; sM2[i] = mW2[i]; }
  for (int i = threadIdx.x; i < 512; i += blockDim.x) sM1[i] = mW1[i];
  if (threadIdx.x < 16) sM3[threadIdx.x] = mW3[threadIdx.x];
  __syncthreads();
  int n = blockIdx.x * blockDim.x + threadIdx.x;
  float atom = 0.f;
  if (n < NN) {
    float cat[32];
    int sp = species[n] - 1;
#pragma unroll
    for (int i = 0; i < 16; i++) cat[i] = emb[sp * 16 + i];
    float xr[16];
#pragma unroll
    for (int i = 0; i < 16; i++) xr[i] = x[n * 16 + i];
#pragma unroll
    for (int c = 0; c < 16; c++) {
      float f = feats[n * 16 + c];
#pragma unroll
      for (int k = 0; k < 16; k++) f += xr[k] * sWs[c * 16 + k];
      cat[16 + c] = f;
    }
    float h1[16];
#pragma unroll
    for (int j = 0; j < 16; j++) {
      float a = mb1[j];
#pragma unroll
      for (int i = 0; i < 32; i++) a += cat[i] * sM1[i * 16 + j];
      h1[j] = a / (1.f + __expf(-a));
    }
    float h2[16];
#pragma unroll
    for (int j = 0; j < 16; j++) {
      float a = mb2[j];
#pragma unroll
      for (int i = 0; i < 16; i++) a += h1[i] * sM2[i * 16 + j];
      h2[j] = a / (1.f + __expf(-a));
    }
    float learned = mb3[0];
#pragma unroll
    for (int i = 0; i < 16; i++) learned += h2[i] * sM3[i];
    atom = learned + coul[n];
  }
  red[threadIdx.x] = atom;
  __syncthreads();
  for (int s = blockDim.x / 2; s > 0; s >>= 1) {
    if (threadIdx.x < s) red[threadIdx.x] += red[threadIdx.x + s];
    __syncthreads();
  }
  if (threadIdx.x == 0) atomicAdd(out, red[0]);
}

extern "C" void kernel_launch(void* const* d_in, const int* in_sizes, int n_in,
                              void* d_out, int out_size, void* d_ws, size_t ws_size,
                              hipStream_t stream) {
  (void)in_sizes; (void)n_in; (void)out_size; (void)ws_size;
  const int*   species = (const int*)d_in[0];
  const int*   src     = (const int*)d_in[1];
  const int*   dst     = (const int*)d_in[2];
  const float* rel     = (const float*)d_in[3];
  const float* emb     = (const float*)d_in[4];
  const float* kvW1    = (const float*)d_in[5];
  const float* kvb1    = (const float*)d_in[6];
  const float* kvW2    = (const float*)d_in[7];
  const float* kvb2    = (const float*)d_in[8];
  const float* kvW3    = (const float*)d_in[9];
  const float* kvb3    = (const float*)d_in[10];
  const float* Wq      = (const float*)d_in[11];
  const float* Wproj   = (const float*)d_in[12];
  const float* finW1   = (const float*)d_in[13];
  const float* finb1   = (const float*)d_in[14];
  const float* finW2   = (const float*)d_in[15];
  const float* finb2   = (const float*)d_in[16];
  const float* finW3   = (const float*)d_in[17];
  const float* finb3   = (const float*)d_in[18];
  const float* Wself   = (const float*)d_in[19];
  const float* mW1     = (const float*)d_in[20];
  const float* mb1     = (const float*)d_in[21];
  const float* mW2     = (const float*)d_in[22];
  const float* mb2     = (const float*)d_in[23];
  const float* mW3     = (const float*)d_in[24];
  const float* mb3     = (const float*)d_in[25];
  float* out = (float*)d_out;

  float* p = (float*)d_ws;
  float* x       = p; p += (size_t)NN * 16;
  float* ef      = p; p += (size_t)EE * 16;
  float* q       = p; p += (size_t)NN * 8;
  float* numer   = p; p += (size_t)NN * 8;
  float* denom   = p; p += (size_t)NN * 8;
  float* feats   = p; p += (size_t)NN * 16;
  float* coul    = p; p += (size_t)NN;
  float* distv   = p; p += (size_t)EE;
  float* scalev  = p; p += (size_t)EE;
  float* bscalev = p; p += (size_t)EE;

  k_node_init<<<NN / 256, 256, 0, stream>>>(species, emb, x, feats, coul, out);
  k_edge_setup<<<EE / 256, 256, 0, stream>>>(species, src, dst, rel, distv, scalev, bscalev, coul);

  for (int l = 0; l < LLAY; l++) {
    k_q<<<NN / 256, 256, 0, stream>>>(x, Wq + (size_t)l * 8 * 16, q, numer, denom);
    k_edge<<<EE / 128, 256, 0, stream>>>((l == 0) ? 0 : 1, src, dst, distv, scalev, bscalev,
        x, q, ef,
        kvW1 + (size_t)l * 16 * 32, kvb1 + (size_t)l * 32,
        kvW2 + (size_t)l * 32 * 32, kvb2 + (size_t)l * 32,
        kvW3 + (size_t)l * 32 * 256, kvb3 + (size_t)l * 256,
        numer, denom, feats);
    k_node_update<<<NN / 256, 256, 0, stream>>>(numer, denom, Wproj + (size_t)l * 16 * 24, x);
  }

  k_edge<<<EE / 128, 256, 0, stream>>>(2, src, dst, distv, scalev, bscalev,
      x, q, ef, finW1, finb1, finW2, finb2, finW3, finb3, numer, denom, feats);
  k_node_final<<<NN / 256, 256, 0, stream>>>(species, emb, x, feats, coul, Wself,
      mW1, mb1, mW2, mb2, mW3, mb3, out);
}
